// MemoryModule_73246372266146
// MI455X (gfx1250) — matrix operand order repaired and checked by reading, not compiled, not run on hardware
//
#include <hip/hip_runtime.h>
#include <hip/hip_bf16.h>

// ---------------------------------------------------------------------------
// TGN memory module on MI455X (gfx1250, wave32, WMMA).
// - GEMMs via v_wmma_f32_16x16x32_f16, fp32->f16 hi/lo split (3 terms/K-chunk)
// - C chains start from inline-0; biases folded into the scalar epilogue
// - node_memory gather + embedding staging via GLOBAL_LOAD_ASYNC_TO_LDS_B128
//   (ASYNCcnt), overlapped with the weight hi/lo swizzle into LDS
// - outputs staged in LDS, streamed out as coalesced float4 stores
// - deterministic last-event-wins scatter via per-node atomicMax in workspace
// ---------------------------------------------------------------------------

typedef __attribute__((ext_vector_type(16))) _Float16 v16h;
typedef __attribute__((ext_vector_type(8)))  float    v8f;
typedef __attribute__((__vector_size__(4 * sizeof(int)))) int v4i;

#define D_DIM        64
#define EV_PER_BLK   128
#define THREADS      256
#define COL_TILES    12           // 192 / 16
#define K_CHUNKS     2            // 64 / 32
// one B plane: [tile][kchunk][lane][16 halfs]
#define PLANE_HALFS  (COL_TILES * K_CHUNKS * 32 * 16)   // 12288
#define PLANE_BYTES  (PLANE_HALFS * 2)                   // 24576

// LDS layout (bytes)
#define OFF_WIH_HI   0
#define OFF_WIH_LO   (OFF_WIH_HI + PLANE_BYTES)
#define OFF_WHH_HI   (OFF_WIH_LO + PLANE_BYTES)
#define OFF_WHH_LO   (OFF_WHH_HI + PLANE_BYTES)
#define OFF_SX       (OFF_WHH_LO + PLANE_BYTES)          // 98304
#define OFF_SH       (OFF_SX + EV_PER_BLK * D_DIM * 4)   // 131072
#define OFF_SDT      (OFF_SH + EV_PER_BLK * D_DIM * 4)   // 163840
#define OFF_SID      (OFF_SDT + EV_PER_BLK * 4)          // 164352
#define SMEM_BYTES   (OFF_SID + EV_PER_BLK * 4)          // 164864 (<320KB/WGP)

// ---- CDNA5 async memory->LDS copy (ASYNCcnt) ------------------------------
#if defined(__has_builtin)
#  if __has_builtin(__builtin_amdgcn_global_load_async_to_lds_b128)
#    define HAVE_ASYNC_BUILTIN 1
#  endif
#  if __has_builtin(__builtin_amdgcn_s_wait_asynccnt)
#    define HAVE_WAIT_ASYNC_BUILTIN 1
#  endif
#endif

typedef __attribute__((address_space(1))) v4i* gv4i_ptr;
typedef __attribute__((address_space(3))) v4i* lv4i_ptr;

__device__ __forceinline__ void async_copy_b128(void* lds_dst, const void* gsrc) {
#if defined(HAVE_ASYNC_BUILTIN)
    __builtin_amdgcn_global_load_async_to_lds_b128(
        (gv4i_ptr)(void*)gsrc,        // global source (addrspace 1, v4i payload)
        (lv4i_ptr)lds_dst,            // LDS destination (addrspace 3)
        /*offset=*/0, /*cpol=*/0);
#else
    unsigned int l  = (unsigned int)(__UINTPTR_TYPE__)lds_dst;
    unsigned long long g = (unsigned long long)(__UINTPTR_TYPE__)gsrc;
    asm volatile("global_load_async_to_lds_b128 %0, %1, off"
                 :: "v"(l), "v"(g) : "memory");
#endif
}

__device__ __forceinline__ void wait_asynccnt0() {
#if defined(HAVE_WAIT_ASYNC_BUILTIN)
    __builtin_amdgcn_s_wait_asynccnt(0);
#else
    asm volatile("s_wait_asynccnt 0" ::: "memory");
#endif
}

// ---------------------------------------------------------------------------
__device__ __forceinline__ v8f wmma_f32_f16(v16h a, v16h b, v8f c) {
    return __builtin_amdgcn_wmma_f32_16x16x32_f16(
        /*neg_a=*/false, a, /*neg_b=*/false, b,
        /*c_mod=*/(short)0, c, /*reuse_a=*/false, /*reuse_b=*/false);
}

// Build A-fragments (16x32 f16, ISA layout) for one 16-event tile from an
// fp32 LDS row block, with hi/lo precision split.
// srow points to sS[(wave*16 + (lane&15)) * 64]; hiHalf = lane>=16.
__device__ __forceinline__ void build_a_frags(const float* __restrict__ srow,
                                              int hiHalf, v16h* fhi, v16h* flo) {
#pragma unroll
    for (int c = 0; c < K_CHUNKS; ++c) {
        v16h ah, al;
#pragma unroll
        for (int v = 0; v < 8; ++v) {
            int kb = ((v & 4) ? 16 : 0) + (hiHalf ? 8 : 0) + ((v & 3) * 2);
            float f0 = srow[c * 32 + kb];
            float f1 = srow[c * 32 + kb + 1];
            _Float16 h0 = (_Float16)f0;
            _Float16 h1 = (_Float16)f1;
            ah[2 * v]     = h0;
            ah[2 * v + 1] = h1;
            al[2 * v]     = (_Float16)(f0 - (float)h0);
            al[2 * v + 1] = (_Float16)(f1 - (float)h1);
        }
        fhi[c] = ah;
        flo[c] = al;
    }
}

__device__ __forceinline__ v16h lds_bfrag(const _Float16* plane, int t, int c, int lane) {
    return *(const v16h*)(plane + (((t * K_CHUNKS + c) * 32 + lane) * 16));
}

// acc += A * B_tile(t) over K=64, 3-term precision split. First call in a
// chain is fed acc == {} so the backend can use the inline-0 SRC2 form.
__device__ __forceinline__ v8f gemm_tile(v8f acc,
                                         const v16h* a_hi, const v16h* a_lo,
                                         const _Float16* plane_hi,
                                         const _Float16* plane_lo,
                                         int t, int lane) {
#pragma unroll
    for (int c = 0; c < K_CHUNKS; ++c) {
        v16h bh = lds_bfrag(plane_hi, t, c, lane);
        v16h bl = lds_bfrag(plane_lo, t, c, lane);
        acc = wmma_f32_f16(a_hi[c], bh, acc);
        acc = wmma_f32_f16(a_lo[c], bh, acc);
        acc = wmma_f32_f16(a_hi[c], bl, acc);
    }
    return acc;
}

__device__ __forceinline__ float sigmoidf_(float x) {
    return 1.0f / (1.0f + __expf(-x));
}

// ---------------------------------------------------------------------------
// Main kernel: gather + time encoding + GRU (WMMA) -> updated_memory
// ---------------------------------------------------------------------------
__global__ __launch_bounds__(THREADS)
void tgn_gru_wmma_kernel(const int*   __restrict__ node_ids,
                         const float* __restrict__ node_embeddings,
                         const float* __restrict__ timestamps,
                         const float* __restrict__ node_memory,
                         const float* __restrict__ last_update_time,
                         const float* __restrict__ W_time,
                         const float* __restrict__ b_time,
                         const float* __restrict__ W_ih,
                         const float* __restrict__ W_hh,
                         const float* __restrict__ b_ih,
                         const float* __restrict__ b_hh,
                         float*       __restrict__ out_updated,
                         int n_events) {
    extern __shared__ char smem[];
    _Float16* sWihHi = (_Float16*)(smem + OFF_WIH_HI);
    _Float16* sWihLo = (_Float16*)(smem + OFF_WIH_LO);
    _Float16* sWhhHi = (_Float16*)(smem + OFF_WHH_HI);
    _Float16* sWhhLo = (_Float16*)(smem + OFF_WHH_LO);
    float*    sX     = (float*)(smem + OFF_SX);
    float*    sH     = (float*)(smem + OFF_SH);
    float*    sDt    = (float*)(smem + OFF_SDT);
    int*      sId    = (int*)(smem + OFF_SID);

    const int tid = threadIdx.x;
    const int e0  = blockIdx.x * EV_PER_BLK;

    // ---- Phase 1: per-event id / dt --------------------------------------
    if (tid < EV_PER_BLK) {
        int e = e0 + tid;
        if (e < n_events) {
            int id   = node_ids[e];
            sId[tid] = id;
            sDt[tid] = timestamps[e] - last_update_time[id];
        } else {
            sId[tid] = 0;
            sDt[tid] = 0.0f;
        }
    }
    __syncthreads();

    // ---- Phase 2: async DMA gather (memory rows) + embeddings into LDS ----
    {
        const float4* emb4 = (const float4*)node_embeddings;
        const float4* mem4 = (const float4*)node_memory;
        for (int i = tid; i < EV_PER_BLK * 16; i += THREADS) {
            int ev = i >> 4;
            int q  = i & 15;
            int e  = e0 + ev;
            int esafe = (e < n_events) ? e : 0;
            async_copy_b128(smem + OFF_SH + (size_t)i * 16,
                            mem4 + (size_t)sId[ev] * 16 + q);
            async_copy_b128(smem + OFF_SX + (size_t)i * 16,
                            emb4 + (size_t)esafe * 16 + q);
        }
    }

    // ---- Phase 3: weight hi/lo swizzle into B-fragment layout -------------
    //      (VALU work overlapping the async DMA above)
    for (int i = tid; i < PLANE_HALFS; i += THREADS) {
        int hv   = i & 15;
        int lane = (i >> 4) & 31;
        int c    = (i >> 9) & 1;
        int t    = i >> 10;
        int j = t * 16 + (lane & 15);                 // output column (gate idx)
        int k = c * 32 + ((hv & 8) ? 16 : 0) + ((lane >> 4) & 1) * 8 + (hv & 7); // K index
        float wi = W_ih[j * D_DIM + k];
        float wh = W_hh[j * D_DIM + k];
        _Float16 wih = (_Float16)wi;
        _Float16 whh = (_Float16)wh;
        sWihHi[i] = wih;
        sWihLo[i] = (_Float16)(wi - (float)wih);
        sWhhHi[i] = whh;
        sWhhLo[i] = (_Float16)(wh - (float)whh);
    }

    wait_asynccnt0();
    __syncthreads();

    // ---- Phase 4: in-place x = emb + dt*W_time + b_time -------------------
    {
        const float4* wt4 = (const float4*)W_time;
        const float4* bt4 = (const float4*)b_time;
        float4* sX4 = (float4*)sX;
        for (int i = tid; i < EV_PER_BLK * 16; i += THREADS) {
            int ev = i >> 4;
            int q  = i & 15;
            float  dt = sDt[ev];
            float4 x  = sX4[i];
            float4 wt = wt4[q];
            float4 bt = bt4[q];
            x.x += dt * wt.x + bt.x;
            x.y += dt * wt.y + bt.y;
            x.z += dt * wt.z + bt.z;
            x.w += dt * wt.w + bt.w;
            sX4[i] = x;
        }
    }
    __syncthreads();

    // ---- Phase 5: per-wave WMMA GEMMs + GRU gate math ---------------------
    const int wave   = tid >> 5;
    const int lane   = tid & 31;
    const int m16    = lane & 15;
    const int hiHalf = lane >> 4;
    const int rowX   = wave * 16 + m16;

    v16h ax_hi[K_CHUNKS], ax_lo[K_CHUNKS];
    v16h ahh_hi[K_CHUNKS], ahh_lo[K_CHUNKS];
    build_a_frags(&sX[rowX * D_DIM], hiHalf, ax_hi, ax_lo);
    build_a_frags(&sH[rowX * D_DIM], hiHalf, ahh_hi, ahh_lo);

    const int jc = lane & 15;

    for (int g = 0; g < 4; ++g) {
        const int tr = g, tz = g + 4, tn = g + 8;

        // bias terms handled in scalar epilogue; C chains start from 0
        float b_r  = b_ih[tr * 16 + jc] + b_hh[tr * 16 + jc];
        float b_z  = b_ih[tz * 16 + jc] + b_hh[tz * 16 + jc];
        float bi_n = b_ih[tn * 16 + jc];
        float bh_n = b_hh[tn * 16 + jc];

        v8f cir = {}, chr = {}, ciz = {}, chz = {}, cin = {}, chn = {};
        cir = gemm_tile(cir, ax_hi,  ax_lo,  sWihHi, sWihLo, tr, lane);
        chr = gemm_tile(chr, ahh_hi, ahh_lo, sWhhHi, sWhhLo, tr, lane);
        ciz = gemm_tile(ciz, ax_hi,  ax_lo,  sWihHi, sWihLo, tz, lane);
        chz = gemm_tile(chz, ahh_hi, ahh_lo, sWhhHi, sWhhLo, tz, lane);
        cin = gemm_tile(cin, ax_hi,  ax_lo,  sWihHi, sWihLo, tn, lane);
        chn = gemm_tile(chn, ahh_hi, ahh_lo, sWhhHi, sWhhLo, tn, lane);

#pragma unroll
        for (int r = 0; r < 8; ++r) {
            int m = r + hiHalf * 8;                 // event row in 16-tile
            int j = g * 16 + jc;                    // output feature column
            float hval = sH[(wave * 16 + m) * D_DIM + j];
            float rg = sigmoidf_(cir[r] + chr[r] + b_r);
            float zg = sigmoidf_(ciz[r] + chz[r] + b_z);
            float ng = tanhf(cin[r] + bi_n + rg * (chn[r] + bh_n));
            float o  = (1.0f - zg) * ng + zg * hval;
            // stage to LDS (sX rows are wave-private; safe to overwrite)
            sX[(wave * 16 + m) * D_DIM + j] = o;
        }
    }
    __syncthreads();

    // ---- Phase 6: coalesced float4 stores of updated_memory ---------------
    {
        const float4* sX4 = (const float4*)sX;
        float4* out4 = (float4*)out_updated;
        for (int i = tid; i < EV_PER_BLK * 16; i += THREADS) {
            int ev = i >> 4;
            int q  = i & 15;
            int e  = e0 + ev;
            if (e < n_events)
                out4[(size_t)e * 16 + q] = sX4[i];
        }
    }
}

// ---------------------------------------------------------------------------
// Scatter support kernels
// ---------------------------------------------------------------------------
__global__ void init_winner_kernel(int* __restrict__ winner, int num_nodes) {
    int i = blockIdx.x * blockDim.x + threadIdx.x;
    if (i < num_nodes) winner[i] = -1;
}

__global__ void vote_winner_kernel(const int* __restrict__ node_ids,
                                   int* __restrict__ winner, int n_events) {
    int i = blockIdx.x * blockDim.x + threadIdx.x;
    if (i < n_events) atomicMax(&winner[node_ids[i]], i);
}

// new_node_memory / new_last_update: copy persistent state, overriding rows
// touched by events with the last (max-index) event's result. Deterministic.
__global__ __launch_bounds__(THREADS)
void finalize_kernel(const int*   __restrict__ winner,
                     const float* __restrict__ updated,
                     const float* __restrict__ node_memory,
                     const float* __restrict__ last_update_time,
                     const float* __restrict__ timestamps,
                     float* __restrict__ out_mem,
                     float* __restrict__ out_last,
                     int num_nodes) {
    int tid  = threadIdx.x;
    int node = blockIdx.x * 16 + (tid >> 4);
    int q    = tid & 15;
    if (node >= num_nodes) return;
    int w = winner[node];
    const float4* src = (w >= 0)
        ? ((const float4*)updated) + (size_t)w * 16
        : ((const float4*)node_memory) + (size_t)node * 16;
    ((float4*)out_mem)[(size_t)node * 16 + q] = src[q];
    if (q == 0)
        out_last[node] = (w >= 0) ? timestamps[w] : last_update_time[node];
}

// Fallback path (no workspace): plain copy + racy scatter (any-wins).
__global__ void copy_state_kernel(const float* __restrict__ node_memory,
                                  const float* __restrict__ last_update_time,
                                  float* __restrict__ out_mem,
                                  float* __restrict__ out_last,
                                  int num_nodes) {
    int i = blockIdx.x * blockDim.x + threadIdx.x;
    if (i < num_nodes * 16)
        ((float4*)out_mem)[i] = ((const float4*)node_memory)[i];
    if (i < num_nodes)
        out_last[i] = last_update_time[i];
}

__global__ void scatter_kernel(const int*   __restrict__ node_ids,
                               const float* __restrict__ updated,
                               const float* __restrict__ timestamps,
                               float* __restrict__ out_mem,
                               float* __restrict__ out_last,
                               int n_events) {
    int i = blockIdx.x * blockDim.x + threadIdx.x;
    if (i >= n_events * 16) return;
    int ev = i >> 4, q = i & 15;
    int id = node_ids[ev];
    ((float4*)out_mem)[(size_t)id * 16 + q] = ((const float4*)updated)[(size_t)ev * 16 + q];
    if (q == 0) out_last[id] = timestamps[ev];
}

// ---------------------------------------------------------------------------
extern "C" void kernel_launch(void* const* d_in, const int* in_sizes, int n_in,
                              void* d_out, int out_size, void* d_ws, size_t ws_size,
                              hipStream_t stream) {
    const int*   node_ids        = (const int*)  d_in[0];
    const float* node_embeddings = (const float*)d_in[1];
    const float* timestamps      = (const float*)d_in[2];
    const float* node_memory     = (const float*)d_in[3];
    const float* last_update     = (const float*)d_in[4];
    const float* W_time          = (const float*)d_in[5];
    const float* b_time          = (const float*)d_in[6];
    const float* W_ih            = (const float*)d_in[7];
    const float* W_hh            = (const float*)d_in[8];
    const float* b_ih            = (const float*)d_in[9];
    const float* b_hh            = (const float*)d_in[10];

    const int n_events  = in_sizes[0];
    const int num_nodes = in_sizes[4];

    float* out_updated = (float*)d_out;
    float* out_mem     = out_updated + (size_t)n_events * D_DIM;
    float* out_last    = out_mem + (size_t)num_nodes * D_DIM;

    // 1) Main WMMA GRU kernel -> updated_memory
    {
        int grid = (n_events + EV_PER_BLK - 1) / EV_PER_BLK;
        tgn_gru_wmma_kernel<<<grid, THREADS, SMEM_BYTES, stream>>>(
            node_ids, node_embeddings, timestamps, node_memory, last_update,
            W_time, b_time, W_ih, W_hh, b_ih, b_hh, out_updated, n_events);
    }

    // 2) Scatter write-back (deterministic last-event-wins if ws fits)
    if (ws_size >= (size_t)num_nodes * sizeof(int)) {
        int* winner = (int*)d_ws;
        init_winner_kernel<<<(num_nodes + THREADS - 1) / THREADS, THREADS, 0, stream>>>(
            winner, num_nodes);
        vote_winner_kernel<<<(n_events + THREADS - 1) / THREADS, THREADS, 0, stream>>>(
            node_ids, winner, n_events);
        finalize_kernel<<<(num_nodes + 15) / 16, THREADS, 0, stream>>>(
            winner, out_updated, node_memory, last_update, timestamps,
            out_mem, out_last, num_nodes);
    } else {
        copy_state_kernel<<<(num_nodes * 16 + THREADS - 1) / THREADS, THREADS, 0, stream>>>(
            node_memory, last_update, out_mem, out_last, num_nodes);
        scatter_kernel<<<(n_events * 16 + THREADS - 1) / THREADS, THREADS, 0, stream>>>(
            node_ids, out_updated, timestamps, out_mem, out_last, n_events);
    }
}
